// PillarNextPillarFeatureNet_10247791969012
// MI455X (gfx1250) — matrix operand order repaired and checked by reading, not compiled, not run on hardware
//
#include <hip/hip_runtime.h>
#include <stdint.h>

typedef __attribute__((ext_vector_type(16))) _Float16 v16h;
typedef __attribute__((ext_vector_type(8)))  float    v8f;
typedef uint32_t u32x4 __attribute__((ext_vector_type(4)));
typedef int      i32x4 __attribute__((ext_vector_type(4)));
typedef int      i32x8 __attribute__((ext_vector_type(8)));

#define GXD 256
#define GYD 256
#define BATCHD 4
#define NKEY  (BATCHD*GXD*GYD + 1)   /* 262145 keys incl. sentinel */
#define NSEGD (BATCHD*GXD*GYD + 2)   /* 262146 segments            */
#define SENTK (BATCHD*GXD*GYD)       /* 262144 sentinel key        */

#define VXC 0.4f
#define VYC 0.4f
#define PX0 (-51.2f)
#define PY0 (-51.2f)
#define BNEPS 1e-3f

// ---------------------------------------------------------------- fills
__global__ void k_zero_f32(float* p, int n) {
  int i = blockIdx.x * blockDim.x + threadIdx.x;
  if (i < n) p[i] = 0.0f;
}

__global__ void k_fill_coords(int* io, int coff, int goff) {
  int i = blockIdx.x * blockDim.x + threadIdx.x;
  if (i < NSEGD) {
    io[coff + i * 3 + 0] = BATCHD;  // sentinel-derived: b=4, cy=0, cx=0
    io[coff + i * 3 + 1] = 0;
    io[coff + i * 3 + 2] = 0;
  }
  if (i == 0) { io[goff + 0] = GYD; io[goff + 1] = GXD; }
}

// ---------------------------------------------------------------- voxelize
__global__ void k_voxelize(const float* __restrict__ pts, int n,
                           int* __restrict__ keyA, float* __restrict__ mval,
                           float* __restrict__ pcnt, float* __restrict__ psum,
                           float* __restrict__ cntv) {
  int i = blockIdx.x * blockDim.x + threadIdx.x;
  bool m = false;
  if (i < n) {
    float b = pts[i * 6 + 0];
    float x = pts[i * 6 + 1];
    float y = pts[i * 6 + 2];
    float z = pts[i * 6 + 3];
    float px = (x - PX0) / VXC;
    float py = (y - PY0) / VYC;
    m = (px >= 0.f) && (px < (float)GXD) && (py >= 0.f) && (py < (float)GYD);
    int cx = (int)px, cy = (int)py;       // trunc, matches astype(int32)
    int bi = (int)b;
    int key = m ? ((bi * GXD + cx) * GYD + cy) : SENTK;
    keyA[i] = key;
    mval[i] = m ? 1.f : 0.f;
    atomicAdd(&pcnt[key], 1.f);
    atomicAdd(&psum[key * 3 + 0], x);
    atomicAdd(&psum[key * 3 + 1], y);
    atomicAdd(&psum[key * 3 + 2], z);
  }
  unsigned long long bal = __ballot(m);
  if ((threadIdx.x & 31) == 0 && bal)
    atomicAdd(cntv, (float)__popcll(bal));
}

// ---------------------------------------------------------------- scan (unique-rank)
__global__ void k_scan1(const float* __restrict__ pcnt, int* __restrict__ rank,
                        int* __restrict__ bsums) {
  __shared__ int sh[256];
  int t = threadIdx.x;
  int base = blockIdx.x * 1024 + t * 4;
  int f[4];
#pragma unroll
  for (int j = 0; j < 4; ++j) {
    int i = base + j;
    f[j] = (i < NKEY && pcnt[i] > 0.f) ? 1 : 0;
  }
  int tsum = f[0] + f[1] + f[2] + f[3];
  sh[t] = tsum;
  __syncthreads();
  for (int off = 1; off < 256; off <<= 1) {
    int v = 0;
    if (t >= off) v = sh[t - off];
    __syncthreads();
    if (t >= off) sh[t] += v;
    __syncthreads();
  }
  int run = sh[t] - tsum;  // exclusive within block
#pragma unroll
  for (int j = 0; j < 4; ++j) {
    int i = base + j;
    if (i < NKEY) rank[i] = run;
    run += f[j];
  }
  if (t == 255) bsums[blockIdx.x] = sh[255];
}

__global__ void k_scan2(int* bsums, int nb) {
  if (threadIdx.x == 0 && blockIdx.x == 0) {
    int run = 0;
    for (int b = 0; b < nb; ++b) { int v = bsums[b]; bsums[b] = run; run += v; }
  }
}

__global__ void k_scan3(int* __restrict__ rank, const int* __restrict__ bsums) {
  int i = blockIdx.x * blockDim.x + threadIdx.x;
  if (i < NKEY) rank[i] += bsums[i >> 10];
}

// ---------------------------------------------------------------- features (f16, A-op K-padded)
__global__ void k_features(const float* __restrict__ pts, int n,
                           const int* __restrict__ keyA,
                           const int* __restrict__ rank,
                           const float* __restrict__ pcnt,
                           const float* __restrict__ psum,
                           int* __restrict__ uinv,
                           _Float16* __restrict__ f16o) {
  int i = blockIdx.x * blockDim.x + threadIdx.x;
  if (i >= n) return;
  int key = keyA[i];
  uinv[i] = rank[key];
  float c = fmaxf(pcnt[key], 1.f);
  float mx = psum[key * 3 + 0] / c;
  float my = psum[key * 3 + 1] / c;
  float mz = psum[key * 3 + 2] / c;
  float x = pts[i * 6 + 1], y = pts[i * 6 + 2], z = pts[i * 6 + 3];
  float it = pts[i * 6 + 4], ts = pts[i * 6 + 5];
  float px = (x - PX0) / VXC, py = (y - PY0) / VYC;
  int cx = (int)px, cy = (int)py;
  float f[10];
  f[0] = x; f[1] = y; f[2] = z; f[3] = it; f[4] = ts;
  f[5] = x - mx; f[6] = y - my; f[7] = z - mz;
  f[8] = x - ((float)cx * VXC + 0.5f * VXC + PX0);
  f[9] = y - ((float)cy * VYC + 0.5f * VYC + PY0);
  _Float16* o = f16o + (size_t)i * 16;
#pragma unroll
  for (int k = 0; k < 10; ++k) o[k] = (_Float16)f[k];
#pragma unroll
  for (int k = 10; k < 16; ++k) o[k] = (_Float16)0.f;
}

// ---------------------------------------------------------------- operand-layout helpers
// A-operand (16x32 f16): lane l holds row m=l&15; lanes>=16 hold K {8..15,24..31}.
// index of (row-in-tile m, K kk in 0..63) inside [tile][kstep][lane][16 halves]:
__device__ __forceinline__ size_t a1_index(size_t tile, int m, int kk) {
  int ks = kk >> 5;
  int k = kk & 31;
  int hi = (k >> 3) & 1;           // 1 -> lanes 16..31
  int kp = k - 8 * hi;             // in {0..7} u {16..23}
  int v = (kp < 8) ? (kp >> 1) : 4 + ((kp - 16) >> 1);
  int h = 2 * v + (kp & 1);
  int lane = m + 16 * hi;
  return ((tile * 2 + ks) * 32 + lane) * 16 + h;
}
// C-operand block: lane holds 8 rows of one column. index of (i, c) given CT col tiles:
__device__ __forceinline__ size_t c_index(int i, int c, int nct) {
  int tile = i >> 4, m = i & 15;
  int ct = c >> 4;
  int lane = (c & 15) + ((m >= 8) ? 16 : 0);
  int r = m & 7;
  return (((size_t)tile * nct + ct) * 32 + lane) * 8 + r;
}

// ---------------------------------------------------------------- weight prep (B-operand order)
// B layout (16-bit 32x16 tile): lane l, element h=2v+j -> K = h+(h>=8?8:0)+(l>=16?8:0), N = (l&15)
__global__ void k_prep_weights(const float* __restrict__ w0,
                               const float* __restrict__ w1,
                               _Float16* __restrict__ w0op,
                               _Float16* __restrict__ w1op) {
  int t = threadIdx.x;
  for (int e = t; e < 2 * 512; e += blockDim.x) {   // w0: 2 col tiles, K padded 10->32
    int op = e >> 9, pos = e & 511;
    int lane = pos >> 4, h = pos & 15;
    int v = h >> 1, j = h & 1;
    int K = 2 * v + j + ((h >= 8) ? 8 : 0) + ((lane >= 16) ? 8 : 0);
    int nc = (lane & 15) + op * 16;
    w0op[e] = (_Float16)((K < 10) ? w0[K * 32 + nc] : 0.f);
  }
  for (int e = t; e < 16 * 512; e += blockDim.x) {  // w1: 8 col tiles x 2 k-steps
    int op = e >> 9, pos = e & 511;
    int ct = op >> 1, ks = op & 1;
    int lane = pos >> 4, h = pos & 15;
    int v = h >> 1, j = h & 1;
    int K = ks * 32 + 2 * v + j + ((h >= 8) ? 8 : 0) + ((lane >= 16) ? 8 : 0);
    int nc = (lane & 15) + ct * 16;
    w1op[e] = (_Float16)w1[K * 128 + nc];
  }
}

// ---------------------------------------------------------------- GEMM0: (Nx10)@(10x32), f16 WMMA
__global__ void k_gemm0(const _Float16* __restrict__ f16i,
                        const _Float16* __restrict__ w0op,
                        const float* __restrict__ mval, int ntiles,
                        _Float16* __restrict__ x0blk,
                        float* __restrict__ sum0, float* __restrict__ ssq0) {
  int gtid = blockIdx.x * blockDim.x + threadIdx.x;
  int wave = gtid >> 5;
  int l = threadIdx.x & 31;
  int nw = (gridDim.x * blockDim.x) >> 5;
  int loff = (l >= 16) ? 8 : 0;
  float s1[2] = {0.f, 0.f}, s2[2] = {0.f, 0.f};
  v16h b0 = *(const v16h*)(w0op + 0 * 512 + l * 16);
  v16h b1 = *(const v16h*)(w0op + 1 * 512 + l * 16);
  for (int tile = wave; tile < ntiles; tile += nw) {
    int row = tile * 16 + (l & 15);
    if (tile + nw < ntiles)
      __builtin_prefetch(f16i + (size_t)(row + nw * 16) * 16, 0, 1);
    union { v16h v; uint32_t u[8]; } a;
#pragma unroll
    for (int v = 0; v < 4; ++v)  // K 0..15 live (K>=10 zero pad), K 16..31 zero
      a.u[v] = *(const uint32_t*)(f16i + (size_t)row * 16 + 2 * v + loff);
#pragma unroll
    for (int v = 4; v < 8; ++v) a.u[v] = 0u;
    float mv[8];
#pragma unroll
    for (int r = 0; r < 8; ++r) mv[r] = mval[tile * 16 + r + loff];

#pragma unroll
    for (int ct = 0; ct < 2; ++ct) {
      v8f acc = {};
      acc = __builtin_amdgcn_wmma_f32_16x16x32_f16(false, a.v, false,
                                                   ct ? b1 : b0, (short)0, acc,
                                                   false, false);
      union { uint4 q; _Float16 h[8]; } p;
#pragma unroll
      for (int r = 0; r < 8; ++r) {
        float y = acc[r];
        p.h[r] = (_Float16)y;
        s1[ct] += mv[r] * y;
        s2[ct] += mv[r] * y * y;
      }
      *(uint4*)(x0blk + (((size_t)tile * 2 + ct) * 32 + l) * 8) = p.q;
    }
  }
#pragma unroll
  for (int ct = 0; ct < 2; ++ct) {
    int col = (l & 15) + ct * 16;
    atomicAdd(&sum0[col], s1[ct]);
    atomicAdd(&ssq0[col], s2[ct]);
  }
}

// ---------------------------------------------------------------- BN scale/shift
__global__ void k_bnstats(const float* __restrict__ sum, const float* __restrict__ ssq,
                          const float* __restrict__ g, const float* __restrict__ b,
                          const float* __restrict__ cntv, int C,
                          float* __restrict__ scale, float* __restrict__ shift) {
  int c = threadIdx.x;
  if (c >= C) return;
  float cnt = fmaxf(cntv[0], 1.f);
  float mu = sum[c] / cnt;
  float var = fmaxf(ssq[c] / cnt - mu * mu, 0.f);
  float sc = g[c] * rsqrtf(var + BNEPS);
  scale[c] = sc;
  shift[c] = b[c] - mu * sc;
}

// ---------------------------------------------------------------- BN+ReLU + seg-max (layer0)
__global__ void k_bnrelu0(const _Float16* __restrict__ x0blk, int n,
                          const float* __restrict__ scale, const float* __restrict__ shift,
                          const int* __restrict__ uinv,
                          _Float16* __restrict__ x1op, float* __restrict__ seg0) {
  long long idx = (long long)blockIdx.x * blockDim.x + threadIdx.x;
  if (idx >= (long long)n * 32) return;
  int i = (int)(idx >> 5), c = (int)(idx & 31);
  float y = (float)x0blk[c_index(i, c, 2)] * scale[c] + shift[c];
  y = fmaxf(y, 0.f);
  x1op[a1_index((size_t)(i >> 4), i & 15, c)] = (_Float16)y;
  atomicMax((unsigned int*)&seg0[(size_t)uinv[i] * 32 + c], __float_as_uint(y));
}

__global__ void k_concat(const float* __restrict__ seg0, const int* __restrict__ uinv,
                         int n, _Float16* __restrict__ x1op) {
  long long idx = (long long)blockIdx.x * blockDim.x + threadIdx.x;
  if (idx >= (long long)n * 32) return;
  int i = (int)(idx >> 5), c = (int)(idx & 31);
  float y = seg0[(size_t)uinv[i] * 32 + c];
  x1op[a1_index((size_t)(i >> 4), i & 15, c + 32)] = (_Float16)y;
}

// ---------------------------------------------------------------- GEMM1: (Nx64)@(64x128), TDM-staged B
__global__ void k_gemm1(const _Float16* __restrict__ x1op,
                        const _Float16* __restrict__ w1op,
                        const float* __restrict__ mval, int ntiles,
                        _Float16* __restrict__ x1blk,
                        float* __restrict__ sum1, float* __restrict__ ssq1) {
  __shared__ __align__(32) uint4 w1s4[1024];  // 16 operand blocks x 512 f16 = 16 KB

#if __has_builtin(__builtin_amdgcn_tensor_load_to_lds) && \
    __has_builtin(__builtin_amdgcn_s_wait_tensorcnt)
  if (threadIdx.x == 0) {
    // D# group0: count=1 | lds_addr | global_addr(57b) | type=2
    uint64_t ga = (uint64_t)(uintptr_t)w1op;
    u32x4 g0;
    g0[0] = 1u;
    g0[1] = (uint32_t)(uintptr_t)&w1s4[0];
    g0[2] = (uint32_t)ga;
    g0[3] = ((uint32_t)(ga >> 32) & 0x01FFFFFFu) | (2u << 30);
    // D# group1: data_size=4B(code 2); 2D tensor 4096x1 dwords, tile 4096x1
    i32x8 g1;
    g1[0] = 2 << 16;              // data_size
    g1[1] = 4096 << 16;           // tensor_dim0[15:0] in [31:16]
    g1[2] = 1 << 16;              // tensor_dim0[31:16]=0 | tensor_dim1[15:0]=1
    g1[3] = 4096 << 16;           // tensor_dim1[31:16]=0 | tile_dim0=4096
    g1[4] = 1;                    // tile_dim1=1, tile_dim2=0
    g1[5] = 4096;                 // tensor_dim0_stride[31:0]
    g1[6] = 0;
    g1[7] = 0;
    i32x4 gz = {0, 0, 0, 0};
#if defined(__clang_major__) && (__clang_major__ >= 23)
    i32x8 gz8 = {0, 0, 0, 0, 0, 0, 0, 0};
    __builtin_amdgcn_tensor_load_to_lds(g0, g1, gz, gz, gz8, 0);
#else
    __builtin_amdgcn_tensor_load_to_lds(g0, g1, gz, gz, 0);
#endif
  }
  if (threadIdx.x < 32) __builtin_amdgcn_s_wait_tensorcnt(0);
#else
  {
    const uint4* src = (const uint4*)w1op;
    for (int i = threadIdx.x; i < 1024; i += blockDim.x) w1s4[i] = src[i];
  }
#endif
  __syncthreads();
  const _Float16* w1s = (const _Float16*)w1s4;

  int gtid = blockIdx.x * blockDim.x + threadIdx.x;
  int wave = gtid >> 5;
  int l = threadIdx.x & 31;
  int nw = (gridDim.x * blockDim.x) >> 5;
  int loff = (l >= 16) ? 8 : 0;
  float s1[8], s2[8];
#pragma unroll
  for (int q = 0; q < 8; ++q) { s1[q] = 0.f; s2[q] = 0.f; }

  for (int tile = wave; tile < ntiles; tile += nw) {
    if (tile + nw < ntiles)  // -> global_prefetch_b8 for next tile's A block
      __builtin_prefetch(x1op + ((size_t)(tile + nw) * 2 * 32 + l) * 16, 0, 1);

    // A operands stored contiguous per (tile, kstep, lane): one 32B load each
    v16h a0 = *(const v16h*)(x1op + (((size_t)tile * 2 + 0) * 32 + l) * 16);
    v16h a1 = *(const v16h*)(x1op + (((size_t)tile * 2 + 1) * 32 + l) * 16);
    float mv[8];
#pragma unroll
    for (int r = 0; r < 8; ++r) mv[r] = mval[tile * 16 + r + loff];

#pragma unroll
    for (int ct = 0; ct < 8; ++ct) {
      v16h b0 = *(const v16h*)(w1s + (ct * 2 + 0) * 512 + l * 16);
      v16h b1 = *(const v16h*)(w1s + (ct * 2 + 1) * 512 + l * 16);
      v8f acc = {};
      acc = __builtin_amdgcn_wmma_f32_16x16x32_f16(false, a0, false, b0,
                                                   (short)0, acc, false, false);
      acc = __builtin_amdgcn_wmma_f32_16x16x32_f16(false, a1, false, b1,
                                                   (short)0, acc, false, false);
      union { uint4 q; _Float16 h[8]; } p;
#pragma unroll
      for (int r = 0; r < 8; ++r) {
        float y = acc[r];
        p.h[r] = (_Float16)y;
        s1[ct] += mv[r] * y;
        s2[ct] += mv[r] * y * y;
      }
      *(uint4*)(x1blk + (((size_t)tile * 8 + ct) * 32 + l) * 8) = p.q;
    }
  }
#pragma unroll
  for (int ct = 0; ct < 8; ++ct) {
    int col = (l & 15) + ct * 16;
    atomicAdd(&sum1[col], s1[ct]);
    atomicAdd(&ssq1[col], s2[ct]);
  }
}

// ---------------------------------------------------------------- BN+ReLU + seg-max into output
__global__ void k_bnrelu1_out(const _Float16* __restrict__ x1blk, int n,
                              const float* __restrict__ scale, const float* __restrict__ shift,
                              const int* __restrict__ uinv, float* __restrict__ feat) {
  long long idx = (long long)blockIdx.x * blockDim.x + threadIdx.x;
  if (idx >= (long long)n * 128) return;
  int i = (int)(idx >> 7), c = (int)(idx & 127);
  float y = (float)x1blk[c_index(i, c, 8)] * scale[c] + shift[c];
  y = fmaxf(y, 0.f);
  atomicMax((unsigned int*)&feat[(size_t)uinv[i] * 128 + c], __float_as_uint(y));
}

// ---------------------------------------------------------------- coords
__global__ void k_coords(const float* __restrict__ pcnt, const int* __restrict__ rank,
                         int* io, int coff) {
  int k = blockIdx.x * blockDim.x + threadIdx.x;
  if (k >= NKEY || !(pcnt[k] > 0.f)) return;
  int pos = rank[k];
  int b = k / (GXD * GYD);
  int rem = k % (GXD * GYD);
  int cx = rem / GYD;
  int cy = rem % GYD;
  io[coff + pos * 3 + 0] = b;
  io[coff + pos * 3 + 1] = cy;
  io[coff + pos * 3 + 2] = cx;
}

// ================================================================ launch
extern "C" void kernel_launch(void* const* d_in, const int* in_sizes, int n_in,
                              void* d_out, int out_size, void* d_ws, size_t ws_size,
                              hipStream_t stream) {
  const float* pts = (const float*)d_in[0];
  const float* w0  = (const float*)d_in[1];
  const float* g0  = (const float*)d_in[2];
  const float* b0  = (const float*)d_in[3];
  const float* w1  = (const float*)d_in[4];
  const float* g1  = (const float*)d_in[5];
  const float* b1  = (const float*)d_in[6];
  const int n = in_sizes[0] / 6;        // 1,000,000 (multiple of 16)
  const int ntiles = n / 16;

  uint8_t* wsb = (uint8_t*)d_ws;
  size_t off = 0;
  auto alloc = [&](size_t bytes) -> void* {
    void* p = wsb + off;
    off = (off + bytes + 255) & ~(size_t)255;
    return p;
  };
  int*      keyA  = (int*)alloc((size_t)n * 4);
  int*      uinv  = (int*)alloc((size_t)n * 4);
  float*    mval  = (float*)alloc((size_t)n * 4);
  float*    pcnt  = (float*)alloc((size_t)NKEY * 4);
  float*    psum  = (float*)alloc((size_t)NKEY * 12);
  int*      rank  = (int*)alloc((size_t)NKEY * 4);
  int*      bsums = (int*)alloc(1024 * 4);
  float*    stats = (float*)alloc(641 * 4);
  float*    cntv = stats + 0;
  float*    sum0 = stats + 1,   *ssq0 = stats + 33;
  float*    sc0  = stats + 65,  *sh0  = stats + 97;
  float*    sum1 = stats + 129, *ssq1 = stats + 257;
  float*    sc1  = stats + 385, *sh1  = stats + 513;
  _Float16* w0op  = (_Float16*)alloc(2 * 512 * 2);
  _Float16* w1op  = (_Float16*)alloc(16 * 512 * 2);
  _Float16* f16f  = (_Float16*)alloc((size_t)n * 16 * 2);
  _Float16* x0blk = (_Float16*)alloc((size_t)n * 32 * 2);
  float*    seg0  = (float*)alloc((size_t)NSEGD * 32 * 4);
  _Float16* x1op  = (_Float16*)alloc((size_t)n * 64 * 2);
  _Float16* x1blk = (_Float16*)alloc((size_t)n * 128 * 2);

  float* fo = (float*)d_out;
  int*   io = (int*)d_out;
  const int coff = NSEGD * 128;
  const int goff = NSEGD * 128 + NSEGD * 3;

  const int B256 = 256;
  auto blks = [](long long items, int bs) { return (unsigned)((items + bs - 1) / bs); };

  // zero accumulators / outputs (graph is replayed -> must re-init every call)
  k_zero_f32<<<blks(NKEY, B256), B256, 0, stream>>>(pcnt, NKEY);
  k_zero_f32<<<blks(NKEY * 3, B256), B256, 0, stream>>>(psum, NKEY * 3);
  k_zero_f32<<<1, B256, 0, stream>>>(stats, 641);
  k_zero_f32<<<blks((long long)NSEGD * 32, B256), B256, 0, stream>>>(seg0, NSEGD * 32);
  k_zero_f32<<<blks((long long)NSEGD * 128, B256), B256, 0, stream>>>(fo, NSEGD * 128);
  k_fill_coords<<<blks(NSEGD, B256), B256, 0, stream>>>(io, coff, goff);

  // voxelize + pillar sums
  k_voxelize<<<blks(n, B256), B256, 0, stream>>>(pts, n, keyA, mval, pcnt, psum, cntv);

  // unique rank (exclusive scan of presence flags)
  int nb1 = (NKEY + 1023) / 1024;
  k_scan1<<<nb1, 256, 0, stream>>>(pcnt, rank, bsums);
  k_scan2<<<1, 1, 0, stream>>>(bsums, nb1);
  k_scan3<<<blks(NKEY, B256), B256, 0, stream>>>(rank, bsums);

  // features + weights
  k_features<<<blks(n, B256), B256, 0, stream>>>(pts, n, keyA, rank, pcnt, psum, uinv, f16f);
  k_prep_weights<<<1, 512, 0, stream>>>(w0, w1, w0op, w1op);

  // layer 0
  k_gemm0<<<512, 256, 0, stream>>>(f16f, w0op, mval, ntiles, x0blk, sum0, ssq0);
  k_bnstats<<<1, 32, 0, stream>>>(sum0, ssq0, g0, b0, cntv, 32, sc0, sh0);
  k_bnrelu0<<<blks((long long)n * 32, B256), B256, 0, stream>>>(x0blk, n, sc0, sh0, uinv, x1op, seg0);
  k_concat<<<blks((long long)n * 32, B256), B256, 0, stream>>>(seg0, uinv, n, x1op);

  // layer 1
  k_gemm1<<<512, 256, 0, stream>>>(x1op, w1op, mval, ntiles, x1blk, sum1, ssq1);
  k_bnstats<<<1, 128, 0, stream>>>(sum1, ssq1, g1, b1, cntv, 128, sc1, sh1);
  k_bnrelu1_out<<<blks((long long)n * 128, B256), B256, 0, stream>>>(x1blk, n, sc1, sh1, uinv, fo);

  // coords
  k_coords<<<blks(NKEY, B256), B256, 0, stream>>>(pcnt, rank, io, coff);

  (void)n_in; (void)out_size; (void)ws_size;
}
